// NKATTransformerLayer_2121713844873
// MI455X (gfx1250) — matrix operand order repaired and checked by reading, not compiled, not run on hardware
//
#include <hip/hip_runtime.h>
#include <hip/hip_bf16.h>

typedef __attribute__((ext_vector_type(16))) _Float16 v16h;
typedef __attribute__((ext_vector_type(8)))  _Float16 v8h;
typedef __attribute__((ext_vector_type(8)))  float    v8f;
typedef __attribute__((ext_vector_type(4)))  float    v4f;
typedef __attribute__((ext_vector_type(4)))  int      i4;

#if defined(__has_builtin)
# if __has_builtin(__builtin_amdgcn_global_load_async_to_lds_b128) && \
     __has_builtin(__builtin_amdgcn_s_wait_asynccnt)
#  define HAVE_ASYNC_LDS 1
# endif
#endif
#ifndef HAVE_ASYNC_LDS
# define HAVE_ASYNC_LDS 0
#endif

#if HAVE_ASYNC_LDS
typedef __attribute__((address_space(1))) i4 gi4;   // global int4
typedef __attribute__((address_space(3))) i4 li4;   // LDS int4
#endif

// ---------------- problem constants ----------------
constexpr int  Dm  = 1024;
constexpr int  Hh  = 16;
constexpr int  DKk = 64;
constexpr int  FFf = 4096;
constexpr int  Bb  = 4;
constexpr int  Ss  = 2048;
constexpr long MROWS  = (long)Bb * Ss;         // 8192
constexpr long XELEMS = MROWS * Dm;            // 8,388,608

// epilogue flags
constexpr int EPI_BIAS   = 1;
constexpr int EPI_GELU   = 2;
constexpr int EPI_RES    = 4;
constexpr int EPI_OUTF16 = 8;
constexpr int EPI_OUTF32 = 16;

// ==========================================================================
// WMMA GEMM: C[M,N] = A[M,K] * B^T, 128x128 tile, BK=32, 256 thr = 8 waves.
//  A: row-major MxK (f16; or f32 converted during staging when AF32).
//  B: row-major N x K always (weights pre-transposed on device).
//  NGUARD: N may not be a multiple of 128 (bounds-check staging + epilogue).
//  When async global->LDS builtins exist and A/B are f16 & unguarded,
//  uses a double-buffered ASYNCcnt pipeline (loads of tile i+1 overlap
//  the 8 WMMAs of tile i).
//  Batched via blockIdx.z: off = (z/HH)*s0 + (z%HH)*s1 per operand.
// ==========================================================================
template<int EPI, bool AF32, bool NGUARD>
__global__ __launch_bounds__(256)
void gemm_wmma(const void* __restrict__ Ap, const _Float16* __restrict__ Bp,
               const float* __restrict__ bias, const float* __restrict__ res,
               float* __restrict__ C32, _Float16* __restrict__ C16,
               int N, int K, int lda, int ldb, int ldc,
               int HH, long sA0, long sA1, long sB0, long sB1,
               long sC0, long sC1)
{
    constexpr int LD = 40;                    // padded K-stride (halves)
    __shared__ __align__(16) _Float16 sA[2][128 * LD];
    __shared__ __align__(16) _Float16 sB[2][128 * LD];

    const int tid  = threadIdx.x;
    const int z    = blockIdx.z;
    const long offA = (long)(z / HH) * sA0 + (long)(z % HH) * sA1;
    const long offB = (long)(z / HH) * sB0 + (long)(z % HH) * sB1;
    const long offC = (long)(z / HH) * sC0 + (long)(z % HH) * sC1;
    const int m0 = blockIdx.y * 128;
    const int n0 = blockIdx.x * 128;

    const _Float16* A16 = (const _Float16*)Ap;
    const float*    A32 = (const float*)Ap;

    const int lane  = tid & 31;
    const int wave  = tid >> 5;
    const int wr    = wave >> 2;              // 0..1 -> 64 rows each
    const int wc    = wave & 3;               // 0..3 -> 32 cols each
    const int laneM = lane & 15;
    const int kbA   = (lane >> 4) << 3;       // A: K {0..7,16..23}/{8..15,24..31}
    const int kbB   = (lane >> 4) << 3;       // B: K {0..7,16..23}/{8..15,24..31}

    v8f acc[8];
    for (int i = 0; i < 8; ++i)
        for (int r = 0; r < 8; ++r) acc[i][r] = 0.f;

    // ---- synchronous staging (fallback / f32-A / guarded-B paths) ----
    auto stageA_sync = [&](int buf, int k0) {
        #pragma unroll
        for (int r2 = 0; r2 < 2; ++r2) {
            int t2 = tid + r2 * 256;
            int ar = t2 >> 2, ac = (t2 & 3) << 3;
            long go = offA + (long)(m0 + ar) * lda + (k0 + ac);
            _Float16* dst = &sA[buf][ar * LD + ac];
            if (AF32) {
                v4f f0 = *(const v4f*)(A32 + go);
                v4f f1 = *(const v4f*)(A32 + go + 4);
                v8h hv;
                #pragma unroll
                for (int i = 0; i < 4; ++i) { hv[i] = (_Float16)f0[i]; hv[i+4] = (_Float16)f1[i]; }
                *(v8h*)dst = hv;
            } else {
                *(v8h*)dst = *(const v8h*)(A16 + go);
            }
        }
    };
    auto stageB_sync = [&](int buf, int k0) {
        #pragma unroll
        for (int r2 = 0; r2 < 2; ++r2) {
            int t2 = tid + r2 * 256;
            int br = t2 >> 2, bc = (t2 & 3) << 3;
            _Float16* dst = &sB[buf][br * LD + bc];
            if (!NGUARD || (n0 + br < N)) {
                long go = offB + (long)(n0 + br) * ldb + (k0 + bc);
                *(v8h*)dst = *(const v8h*)(Bp + go);
            } else {
                v8h zz;
                #pragma unroll
                for (int i = 0; i < 8; ++i) zz[i] = (_Float16)0.f;
                *(v8h*)dst = zz;
            }
        }
    };

    // ---- compute: 4x2 fragments, 8 WMMAs, per wave ----
    auto compute = [&](int buf) {
        v16h af[4], bf[2];
        #pragma unroll
        for (int mt = 0; mt < 4; ++mt) {
            int base = (wr * 64 + mt * 16 + laneM) * LD + kbA;
            v8h lo = *(const v8h*)&sA[buf][base];
            v8h hi = *(const v8h*)&sA[buf][base + 16];
            af[mt] = __builtin_shufflevector(lo, hi, 0,1,2,3,4,5,6,7,8,9,10,11,12,13,14,15);
        }
        #pragma unroll
        for (int nt = 0; nt < 2; ++nt) {
            int base = (wc * 32 + nt * 16 + laneM) * LD + kbB;
            v8h lo = *(const v8h*)&sB[buf][base];
            v8h hi = *(const v8h*)&sB[buf][base + 16];
            bf[nt] = __builtin_shufflevector(lo, hi, 0,1,2,3,4,5,6,7,8,9,10,11,12,13,14,15);
        }
        #pragma unroll
        for (int mt = 0; mt < 4; ++mt)
            #pragma unroll
            for (int nt = 0; nt < 2; ++nt)
                acc[mt * 2 + nt] = __builtin_amdgcn_wmma_f32_16x16x32_f16(
                    false, af[mt], false, bf[nt], (short)0, acc[mt * 2 + nt],
                    false, false);
    };

#if HAVE_ASYNC_LDS
    constexpr bool ASYNCOK = !AF32 && !NGUARD;
    if constexpr (ASYNCOK) {
        // issues exactly 4 wave-wide async instructions per 32-K tile
        auto stage_async = [&](int buf, int k0) {
            #pragma unroll
            for (int r2 = 0; r2 < 2; ++r2) {
                int t2 = tid + r2 * 256;
                int ar = t2 >> 2, ac = (t2 & 3) << 3;
                long go = offA + (long)(m0 + ar) * lda + (k0 + ac);
                __builtin_amdgcn_global_load_async_to_lds_b128(
                    (gi4*)(A16 + go), (li4*)&sA[buf][ar * LD + ac], 0, 0);
            }
            #pragma unroll
            for (int r2 = 0; r2 < 2; ++r2) {
                int t2 = tid + r2 * 256;
                int br = t2 >> 2, bc = (t2 & 3) << 3;
                long go = offB + (long)(n0 + br) * ldb + (k0 + bc);
                __builtin_amdgcn_global_load_async_to_lds_b128(
                    (gi4*)(Bp + go), (li4*)&sB[buf][br * LD + bc], 0, 0);
            }
        };
        stage_async(0, 0);
        int buf = 0;
        for (int k0 = 0; k0 < K; k0 += 32) {
            if (k0 + 32 < K) {
                stage_async(buf ^ 1, k0 + 32);
                __builtin_amdgcn_s_wait_asynccnt(4);   // current tile landed
            } else {
                __builtin_amdgcn_s_wait_asynccnt(0);
            }
            __syncthreads();
            compute(buf);
            __syncthreads();                            // WAR before reuse
            buf ^= 1;
        }
    } else
#endif
    {
        for (int k0 = 0; k0 < K; k0 += 32) {
            stageA_sync(0, k0);
            stageB_sync(0, k0);
            __syncthreads();
            compute(0);
            __syncthreads();
        }
    }

    // ---- epilogue ----
    #pragma unroll
    for (int mt = 0; mt < 4; ++mt)
        #pragma unroll
        for (int nt = 0; nt < 2; ++nt) {
            v8f c = acc[mt * 2 + nt];
            int rbase = m0 + wr * 64 + mt * 16 + ((lane >> 4) << 3);
            int col   = n0 + wc * 32 + nt * 16 + laneM;
            if (!NGUARD || col < N) {
                float bv = (EPI & EPI_BIAS) ? bias[col] : 0.f;
                #pragma unroll
                for (int r = 0; r < 8; ++r) {
                    float v = c[r] + bv;
                    if (EPI & EPI_GELU) v = 0.5f * v * (1.f + erff(v * 0.70710678118f));
                    long ci = offC + (long)(rbase + r) * ldc + col;
                    if (EPI & EPI_RES)    v += res[ci];
                    if (EPI & EPI_OUTF32) C32[ci] = v;
                    if (EPI & EPI_OUTF16) C16[ci] = (_Float16)v;
                }
            }
        }
}

// ==========================================================================
// fp32 [K][N] weight -> f16 [N][K], LDS-tiled 32x32 transpose
// ==========================================================================
__global__ __launch_bounds__(256)
void cvt_transpose(const float* __restrict__ in, _Float16* __restrict__ out,
                   int K, int N)
{
    __shared__ _Float16 tile[32][33];
    const int tx = threadIdx.x, ty = threadIdx.y;
    const int kt = blockIdx.y * 32, nt = blockIdx.x * 32;
    #pragma unroll
    for (int j = 0; j < 4; ++j)
        tile[ty + j * 8][tx] = (_Float16)in[(long)(kt + ty + j * 8) * N + nt + tx];
    __syncthreads();
    #pragma unroll
    for (int j = 0; j < 4; ++j)
        out[(long)(nt + ty + j * 8) * K + kt + tx] = tile[tx][ty + j * 8];
}

// V [b,s,H*64] f16 -> Vt [b,h,64,S] f16 (per-head transpose)
__global__ __launch_bounds__(256)
void vhead_transpose(const _Float16* __restrict__ v, _Float16* __restrict__ vt)
{
    __shared__ _Float16 tile[32][33];
    const int tx = threadIdx.x, ty = threadIdx.y;
    const int b = blockIdx.z >> 4, h = blockIdx.z & 15;
    const int d0 = blockIdx.x * 32, s0 = blockIdx.y * 32;
    #pragma unroll
    for (int j = 0; j < 4; ++j)
        tile[ty + j * 8][tx] = v[((long)b * Ss + s0 + ty + j * 8) * Dm + h * DKk + d0 + tx];
    __syncthreads();
    #pragma unroll
    for (int j = 0; j < 4; ++j)
        vt[(((long)b * Hh + h) * DKk + d0 + ty + j * 8) * Ss + s0 + tx] = tile[tx][ty + j * 8];
}

// ==========================================================================
// LayerNorm over D=1024; optional ||y|| row-norm output (for cwgt).
// ==========================================================================
__global__ __launch_bounds__(256)
void ln_rows(const float* __restrict__ x, const float* __restrict__ g,
             const float* __restrict__ b, _Float16* __restrict__ out16,
             float* __restrict__ norms, int withNorm)
{
    __shared__ float red[256];
    const int row = blockIdx.x, t = threadIdx.x;
    const float* xr = x + (long)row * Dm;
    float v[4], s = 0.f;
    #pragma unroll
    for (int j = 0; j < 4; ++j) { v[j] = xr[t + j * 256]; s += v[j]; }
    red[t] = s; __syncthreads();
    for (int o = 128; o > 0; o >>= 1) { if (t < o) red[t] += red[t + o]; __syncthreads(); }
    float mu = red[0] * (1.f / Dm); __syncthreads();
    float vs = 0.f;
    #pragma unroll
    for (int j = 0; j < 4; ++j) { float d = v[j] - mu; vs += d * d; }
    red[t] = vs; __syncthreads();
    for (int o = 128; o > 0; o >>= 1) { if (t < o) red[t] += red[t + o]; __syncthreads(); }
    float rstd = rsqrtf(red[0] * (1.f / Dm) + 1e-5f); __syncthreads();
    float ns = 0.f;
    #pragma unroll
    for (int j = 0; j < 4; ++j) {
        int d = t + j * 256;
        float y = (v[j] - mu) * rstd * g[d] + b[d];
        out16[(long)row * Dm + d] = (_Float16)y;
        ns += y * y;
    }
    if (withNorm) {
        red[t] = ns; __syncthreads();
        for (int o = 128; o > 0; o >>= 1) { if (t < o) red[t] += red[t + o]; __syncthreads(); }
        if (t == 0) norms[row] = sqrtf(red[0]);
    }
}

// row softmax over 2048 logits, scaled by 1/sqrt(DK)=0.125, in place.
__global__ __launch_bounds__(256)
void softmax_rows(float* __restrict__ attn)
{
    __shared__ float red[256];
    const long row = blockIdx.x; const int t = threadIdx.x;
    float* p = attn + row * (long)Ss;
    float v[8], m = -1e30f;
    #pragma unroll
    for (int j = 0; j < 8; ++j) { v[j] = p[t + j * 256] * 0.125f; m = fmaxf(m, v[j]); }
    red[t] = m; __syncthreads();
    for (int o = 128; o > 0; o >>= 1) { if (t < o) red[t] = fmaxf(red[t], red[t + o]); __syncthreads(); }
    m = red[0]; __syncthreads();
    float s = 0.f;
    #pragma unroll
    for (int j = 0; j < 8; ++j) { v[j] = expf(v[j] - m); s += v[j]; }
    red[t] = s; __syncthreads();
    for (int o = 128; o > 0; o >>= 1) { if (t < o) red[t] += red[t + o]; __syncthreads(); }
    float inv = 1.f / red[0];
    #pragma unroll
    for (int j = 0; j < 8; ++j) p[t + j * 256] = v[j] * inv;
}

// Qg = Q*cos(phase[h]) - V*sin(phase[h]), h = (d%D)/64
__global__ void gauge_q(const _Float16* __restrict__ q, const _Float16* __restrict__ v,
                        const float* __restrict__ ph, _Float16* __restrict__ qg, long n)
{
    long i = (long)blockIdx.x * blockDim.x + threadIdx.x;
    if (i >= n) return;
    int h = (int)((i & (Dm - 1)) >> 6);
    float c = cosf(ph[h]), s = sinf(ph[h]);
    qg[i] = (_Float16)((float)q[i] * c - (float)v[i] * s);
}

// x2 = x1 + beta*h2 + exp(-alpha*||h2||_row)*t
__global__ void combine_conv(const float* __restrict__ x1, const _Float16* __restrict__ h2,
                             const float* __restrict__ t, const float* __restrict__ norms,
                             const float* __restrict__ alpha, const float* __restrict__ beta,
                             float* __restrict__ x2, long n)
{
    long i = (long)blockIdx.x * blockDim.x + threadIdx.x;
    if (i >= n) return;
    long row = i >> 10;
    float cw = expf(-alpha[0] * norms[row]);
    x2[i] = x1[i] + beta[0] * (float)h2[i] + cw * t[i];
}

// ==========================================================================
extern "C" void kernel_launch(void* const* d_in, const int* in_sizes, int n_in,
                              void* d_out, int out_size, void* d_ws, size_t ws_size,
                              hipStream_t stream)
{
    (void)in_sizes; (void)n_in; (void)out_size; (void)ws_size;
    const float* x0   = (const float*)d_in[0];
    const float* wq   = (const float*)d_in[1];
    const float* wk   = (const float*)d_in[2];
    const float* wv   = (const float*)d_in[3];
    const float* wo   = (const float*)d_in[4];
    const float* gph  = (const float*)d_in[5];
    const float* cw1  = (const float*)d_in[6];
    const float* cb1  = (const float*)d_in[7];
    const float* cw2  = (const float*)d_in[8];
    const float* cb2  = (const float*)d_in[9];
    const float* alpha= (const float*)d_in[10];
    const float* beta = (const float*)d_in[11];
    const float* fw1  = (const float*)d_in[12];
    const float* fb1  = (const float*)d_in[13];
    const float* fw2  = (const float*)d_in[14];
    const float* fb2  = (const float*)d_in[15];
    const float* ln1g = (const float*)d_in[16];
    const float* ln1b = (const float*)d_in[17];
    const float* ln2g = (const float*)d_in[18];
    const float* ln2b = (const float*)d_in[19];
    const float* ln3g = (const float*)d_in[20];
    const float* ln3b = (const float*)d_in[21];

    float* outx = (float*)d_out;
    float* attn = outx + XELEMS;

    size_t cur = 0;
    auto a16 = [&](long n) { _Float16* p = (_Float16*)((char*)d_ws + cur);
                             cur = (cur + (size_t)n * 2 + 255) & ~(size_t)255; return p; };
    auto a32 = [&](long n) { float* p = (float*)((char*)d_ws + cur);
                             cur = (cur + (size_t)n * 4 + 255) & ~(size_t)255; return p; };

    _Float16 *wqt = a16(Dm*Dm), *wkt = a16(Dm*Dm), *wvt = a16(Dm*Dm), *wot = a16(Dm*Dm);
    _Float16 *cw1t = a16((long)Dm*2*Dm), *cw2t = a16((long)2*Dm*Dm);
    _Float16 *fw1t = a16((long)Dm*FFf),  *fw2t = a16((long)FFf*Dm);
    _Float16 *h16  = a16(XELEMS), *Q16 = a16(XELEMS), *K16 = a16(XELEMS), *V16 = a16(XELEMS);
    _Float16 *Qg16 = a16(XELEMS), *Vt = a16(XELEMS), *ctx16 = a16(XELEMS);
    _Float16 *h2_16 = a16(XELEMS), *h3_16 = a16(XELEMS);
    _Float16 *t1 = a16(MROWS * 2L*Dm), *f1 = a16(MROWS * (long)FFf);
    float *x1 = a32(XELEMS), *tt = a32(XELEMS), *x2 = a32(XELEMS), *norms = a32(MROWS);

    // ---- weight down-convert + transpose (fp32 KxN -> f16 NxK) ----
    const dim3 tb(32, 8);
    cvt_transpose<<<dim3(Dm/32,  Dm/32),   tb, 0, stream>>>(wq,  wqt,  Dm,   Dm);
    cvt_transpose<<<dim3(Dm/32,  Dm/32),   tb, 0, stream>>>(wk,  wkt,  Dm,   Dm);
    cvt_transpose<<<dim3(Dm/32,  Dm/32),   tb, 0, stream>>>(wv,  wvt,  Dm,   Dm);
    cvt_transpose<<<dim3(Dm/32,  Dm/32),   tb, 0, stream>>>(wo,  wot,  Dm,   Dm);
    cvt_transpose<<<dim3(2*Dm/32, Dm/32),  tb, 0, stream>>>(cw1, cw1t, Dm,   2*Dm);
    cvt_transpose<<<dim3(Dm/32,  2*Dm/32), tb, 0, stream>>>(cw2, cw2t, 2*Dm, Dm);
    cvt_transpose<<<dim3(FFf/32, Dm/32),   tb, 0, stream>>>(fw1, fw1t, Dm,   FFf);
    cvt_transpose<<<dim3(Dm/32,  FFf/32),  tb, 0, stream>>>(fw2, fw2t, FFf,  Dm);

    const long SD = (long)Ss * Dm, SS = (long)Ss * Ss;

    // ---- LN1 -> h ----
    ln_rows<<<(unsigned)MROWS, 256, 0, stream>>>(x0, ln1g, ln1b, h16, nullptr, 0);

    // ---- Q,K,V projections (M=8192,N=1024,K=1024) ----
    dim3 gP(Dm/128, (unsigned)(MROWS/128), 1);
    gemm_wmma<EPI_OUTF16,false,false><<<gP,256,0,stream>>>(h16, wqt, nullptr, nullptr,
        nullptr, Q16, Dm, Dm, Dm, Dm, Dm, 1, 0,0,0,0,0,0);
    gemm_wmma<EPI_OUTF16,false,false><<<gP,256,0,stream>>>(h16, wkt, nullptr, nullptr,
        nullptr, K16, Dm, Dm, Dm, Dm, Dm, 1, 0,0,0,0,0,0);
    gemm_wmma<EPI_OUTF16,false,false><<<gP,256,0,stream>>>(h16, wvt, nullptr, nullptr,
        nullptr, V16, Dm, Dm, Dm, Dm, Dm, 1, 0,0,0,0,0,0);

    // ---- gauge-modified query; V per-head transpose ----
    gauge_q<<<(unsigned)(XELEMS/256), 256, 0, stream>>>(Q16, V16, gph, Qg16, XELEMS);
    vhead_transpose<<<dim3(DKk/32, Ss/32, Bb*Hh), tb, 0, stream>>>(V16, Vt);

    // ---- scores = Qg . K^T per (b,h): M=N=2048, K=64 -> d_out attn (raw) ----
    gemm_wmma<EPI_OUTF32,false,false><<<dim3(Ss/128, Ss/128, Bb*Hh),256,0,stream>>>(
        Qg16, K16, nullptr, nullptr, attn, nullptr,
        Ss, DKk, Dm, Dm, Ss, Hh, SD, DKk, SD, DKk, (long)Hh*SS, SS);

    // ---- softmax rows (scale 1/8) in place: this IS the attn output ----
    softmax_rows<<<(unsigned)(Bb*Hh*Ss), 256, 0, stream>>>(attn);

    // ---- ctx = P . V per (b,h): M=2048, N=64, K=2048 (A fp32 from d_out) ----
    gemm_wmma<EPI_OUTF16,true,true><<<dim3(1, Ss/128, Bb*Hh),256,0,stream>>>(
        attn, Vt, nullptr, nullptr, nullptr, ctx16,
        DKk, Ss, Ss, Ss, Dm, Hh, (long)Hh*SS, SS,
        (long)Hh*DKk*Ss, (long)DKk*Ss, SD, DKk);

    // ---- x1 = x0 + ctx . wo ----
    gemm_wmma<EPI_RES|EPI_OUTF32,false,false><<<gP,256,0,stream>>>(ctx16, wot,
        nullptr, x0, x1, nullptr, Dm, Dm, Dm, Dm, Dm, 1, 0,0,0,0,0,0);

    // ---- superconvergence block ----
    ln_rows<<<(unsigned)MROWS, 256, 0, stream>>>(x1, ln2g, ln2b, h2_16, norms, 1);
    gemm_wmma<EPI_BIAS|EPI_GELU|EPI_OUTF16,false,false>
        <<<dim3(2*Dm/128, (unsigned)(MROWS/128), 1),256,0,stream>>>(h2_16, cw1t, cb1,
        nullptr, nullptr, t1, 2*Dm, Dm, Dm, Dm, 2*Dm, 1, 0,0,0,0,0,0);
    gemm_wmma<EPI_BIAS|EPI_OUTF32,false,false><<<gP,256,0,stream>>>(t1, cw2t, cb2,
        nullptr, tt, nullptr, Dm, 2*Dm, 2*Dm, 2*Dm, Dm, 1, 0,0,0,0,0,0);
    combine_conv<<<(unsigned)(XELEMS/256), 256, 0, stream>>>(x1, h2_16, tt, norms,
        alpha, beta, x2, XELEMS);

    // ---- feed-forward ----
    ln_rows<<<(unsigned)MROWS, 256, 0, stream>>>(x2, ln3g, ln3b, h3_16, nullptr, 0);
    gemm_wmma<EPI_BIAS|EPI_GELU|EPI_OUTF16,false,false>
        <<<dim3(FFf/128, (unsigned)(MROWS/128), 1),256,0,stream>>>(h3_16, fw1t, fb1,
        nullptr, nullptr, f1, FFf, Dm, Dm, Dm, FFf, 1, 0,0,0,0,0,0);
    gemm_wmma<EPI_BIAS|EPI_RES|EPI_OUTF32,false,false><<<gP,256,0,stream>>>(f1, fw2t, fb2,
        x2, outx, nullptr, Dm, FFf, FFf, FFf, Dm, 1, 0,0,0,0,0,0);
}
